// AttnBlock_68006512165205
// MI455X (gfx1250) — matrix operand order repaired and checked by reading, not compiled, not run on hardware
//
#include <hip/hip_runtime.h>
#include <stdint.h>

// ---------------------------------------------------------------------------
// CDNA5 (gfx1250) wave32 WMMA attention block.
//   x:(8,64,64,64) f32 -> GN(32 groups) -> Q/K/V linears(last axis) ->
//   flash attention over 4096 tokens/batch (dim 64) -> proj linear -> +res
// All matmuls: v_wmma_f32_16x16x32_f16 (f16 operands, f32 accum).
// ---------------------------------------------------------------------------

typedef __attribute__((ext_vector_type(16))) _Float16 v16h;
typedef __attribute__((ext_vector_type(8)))  float    v8f;
typedef __attribute__((ext_vector_type(4)))  int      i4;

union AB16 { v16h v; uint32_t u[8]; _Float16 h[16]; };
union ACC8 { v8f  v; float    f[8]; };

#define WMMA16(A,B,Cc) __builtin_amdgcn_wmma_f32_16x16x32_f16(false,(A),false,(B),(short)0,(Cc),false,false)

// A-operand (16x32 f16) VGPR i holds K pair: base(i) = hi*16 + 2i
__device__ __forceinline__ int a_kbase(int i, int hi) {
  return hi*16 + 2*i;
}
// B-operand (32x16 f16) VGPR i holds K pair: hi*16 + 2i ; column N = lane&15

// --- async-to-LDS probe (gfx1250 GLOBAL_LOAD_ASYNC_TO_LDS_B128, ASYNCcnt) ---
#if defined(__has_builtin)
#if __has_builtin(__builtin_amdgcn_global_load_async_to_lds_b128)
#define HAVE_ASYNC_LDS 1
#endif
#endif
#ifndef HAVE_ASYNC_LDS
#define HAVE_ASYNC_LDS 0
#endif

__device__ __forceinline__ void copy16_to_lds(const void* g, void* l) {
#if HAVE_ASYNC_LDS
  // signature (from probe diagnostic): (int4 __device__* gsrc, int4* lds, imm off, imm cpol)
  __builtin_amdgcn_global_load_async_to_lds_b128((i4*)g, (i4*)l, 0, 0);
#else
  *(uint4*)l = *(const uint4*)g;
#endif
}
__device__ __forceinline__ void async_fence() {
#if HAVE_ASYNC_LDS
#if defined(__has_builtin) && __has_builtin(__builtin_amdgcn_s_wait_asynccnt)
  __builtin_amdgcn_s_wait_asynccnt(0);
#else
  asm volatile("s_wait_asynccnt 0" ::: "memory");
#endif
#endif
}

// ---------------------------------------------------------------------------
// Kernel 0: convert the four 64x64 fp32 weight matrices to f16 once.
//   wh layout: [Wq | Wk | Wv | Wp], each 4096 f16.
// ---------------------------------------------------------------------------
__global__ __launch_bounds__(256) void prep_weights(
    const float* __restrict__ Wq, const float* __restrict__ Wk,
    const float* __restrict__ Wv, const float* __restrict__ Wp,
    _Float16* __restrict__ wh) {
  int idx = blockIdx.x * 256 + threadIdx.x;   // 16 blocks * 256 = 4096
  wh[idx]           = (_Float16)Wq[idx];
  wh[4096 + idx]    = (_Float16)Wk[idx];
  wh[8192 + idx]    = (_Float16)Wv[idx];
  wh[12288 + idx]   = (_Float16)Wp[idx];
}

// ---------------------------------------------------------------------------
// Kernel 1: GroupNorm(32 groups over C=64) -> xn (f16, natural b,c,h,w)
// one block per (b,g); group slice = 2 channels * 4096 = 8192 contiguous f32
// ---------------------------------------------------------------------------
__global__ __launch_bounds__(256) void gn_kernel(const float* __restrict__ x,
                                                 const float* __restrict__ gw,
                                                 const float* __restrict__ gb,
                                                 _Float16* __restrict__ xn) {
  __shared__ float r1[256], r2[256];
  int b = blockIdx.x >> 5, g = blockIdx.x & 31;
  size_t base = ((size_t)(b*64 + g*2)) * 4096;
  float s = 0.f, ss = 0.f;
  for (int i = threadIdx.x; i < 8192; i += 256) {
    float v = x[base + i]; s += v; ss += v*v;
  }
  r1[threadIdx.x] = s; r2[threadIdx.x] = ss;
  __syncthreads();
  for (int off = 128; off > 0; off >>= 1) {
    if ((int)threadIdx.x < off) {
      r1[threadIdx.x] += r1[threadIdx.x + off];
      r2[threadIdx.x] += r2[threadIdx.x + off];
    }
    __syncthreads();
  }
  float mu   = r1[0] * (1.f/8192.f);
  float var  = r2[0] * (1.f/8192.f) - mu*mu;
  float rinv = rsqrtf(var + 1e-5f);
  for (int i = threadIdx.x; i < 8192; i += 256) {
    int c = g*2 + (i >> 12);
    float v = (x[base + i] - mu) * rinv * gw[c] + gb[c];
    xn[base + i] = (_Float16)v;
  }
}

// ---------------------------------------------------------------------------
// One 64x64 GEMM phase vs an f16 weight matrix; stages result into LDS y.
//   TOKEN_MAJOR: y[j][c] (for token-major q/k stores); else y[c][j] (for v).
// ---------------------------------------------------------------------------
template<bool TOKEN_MAJOR>
__device__ __forceinline__ void gemm_phase(const AB16& a0, const AB16& a1,
    const _Float16* __restrict__ Wh, const float* __restrict__ bias,
    _Float16* y, int wave, int lo, int hi) {
  #pragma unroll
  for (int jt = 0; jt < 4; ++jt) {
    int j = jt*16 + lo;
    AB16 b0, b1;                      // B[K=w][N=j] = W[j][w]; pairs contiguous
    #pragma unroll
    for (int i = 0; i < 8; ++i) {
      int c0 = hi*16 + 2*i;
      b0.u[i] = *(const uint32_t*)(Wh + j*64 + c0);
      b1.u[i] = *(const uint32_t*)(Wh + j*64 + 32 + c0);
    }
    ACC8 acc;
    #pragma unroll
    for (int e = 0; e < 8; ++e) acc.f[e] = 0.f;
    acc.v = WMMA16(a0.v, b0.v, acc.v);
    acc.v = WMMA16(a1.v, b1.v, acc.v);
    float bj = bias[j];
    #pragma unroll
    for (int i = 0; i < 8; ++i) {
      int c = wave*16 + i + 8*hi;
      if (TOKEN_MAJOR) y[j*64 + c] = (_Float16)(acc.f[i] + bj);
      else             y[c*64 + j] = (_Float16)(acc.f[i] + bj);
    }
  }
}

// ---------------------------------------------------------------------------
// Kernel 2: Q/K/V linears on last axis.  One block per (b,h): 64 rows (c) x 64.
// Outputs: qT,kT token-major [b][n][c] (n = h*64+j); vN channel-major [b][c][n].
// ---------------------------------------------------------------------------
__global__ __launch_bounds__(128) void qkv_kernel(const _Float16* __restrict__ xn,
    const _Float16* __restrict__ wh,
    const float* __restrict__ bq, const float* __restrict__ bk,
    const float* __restrict__ bv,
    _Float16* __restrict__ qT, _Float16* __restrict__ kT, _Float16* __restrict__ vN) {
  __shared__ __align__(16) _Float16 y[64*64];
  int b = blockIdx.x >> 6, h = blockIdx.x & 63;
  int lane = threadIdx.x & 31, wave = threadIdx.x >> 5;
  int lo = lane & 15, hi = lane >> 4;

  // A operand: rows = channels c = wave*16+lo, K = w (contiguous pairs)
  const _Float16* xrow = xn + (((size_t)(b*64 + wave*16 + lo))*64 + h)*64;
  AB16 a0, a1;
  #pragma unroll
  for (int i = 0; i < 8; ++i) {
    int kb = a_kbase(i, hi);
    a0.u[i] = *(const uint32_t*)(xrow + kb);
    a1.u[i] = *(const uint32_t*)(xrow + kb + 32);
  }

  // ---- Q ----
  gemm_phase<true >(a0, a1, wh,          bq, y, wave, lo, hi);
  __syncthreads();
  {
    uint4* ds = (uint4*)(qT + ((size_t)b*4096 + h*64)*64);  // contiguous block
    const uint4* ys = (const uint4*)y;
    for (int u = threadIdx.x; u < 512; u += 128) ds[u] = ys[u];
  }
  __syncthreads();
  // ---- K ----
  gemm_phase<true >(a0, a1, wh + 4096,   bk, y, wave, lo, hi);
  __syncthreads();
  {
    uint4* ds = (uint4*)(kT + ((size_t)b*4096 + h*64)*64);
    const uint4* ys = (const uint4*)y;
    for (int u = threadIdx.x; u < 512; u += 128) ds[u] = ys[u];
  }
  __syncthreads();
  // ---- V (channel-major rows of 128B) ----
  gemm_phase<false>(a0, a1, wh + 8192,   bv, y, wave, lo, hi);
  __syncthreads();
  {
    const uint4* ys = (const uint4*)y;
    for (int u = threadIdx.x; u < 512; u += 128) {
      int c = u >> 3, q4 = u & 7;
      ((uint4*)(vN + ((size_t)(b*64 + c))*4096 + h*64))[q4] = ys[u];
    }
  }
}

// ---------------------------------------------------------------------------
// Kernel 3: flash attention, double-buffered K/V tiles in LDS.
// One block per (b, 64-query block); 4 waves, each owns a 16-query tile.
// ---------------------------------------------------------------------------
__global__ __launch_bounds__(128) void attn_kernel(const _Float16* __restrict__ qT,
    const _Float16* __restrict__ kT, const _Float16* __restrict__ vN,
    _Float16* __restrict__ attn) {
  __shared__ __align__(16) _Float16 s_q[64*64];      // Q block / P scratch / O staging
  __shared__ __align__(16) _Float16 s_k[2][64*64];   // [token][c]
  __shared__ __align__(16) _Float16 s_v[2][64*64];   // [c][token]
  int b = blockIdx.x >> 6, qb = blockIdx.x & 63;
  int qbase = qb * 64;
  int lane = threadIdx.x & 31, wave = threadIdx.x >> 5;
  int lo = lane & 15, hi = lane >> 4;

  const _Float16* kbp = kT + ((size_t)b*4096)*64;
  const _Float16* vbp = vN + ((size_t)b*64)*4096;

  // stage Q block + first K/V tiles
  {
    const uint4* src = (const uint4*)(qT + ((size_t)b*4096 + qbase)*64);
    for (int u = threadIdx.x; u < 512; u += 128)
      copy16_to_lds(src + u, (uint4*)s_q + u);
    const uint4* ks = (const uint4*)kbp;
    for (int u = threadIdx.x; u < 512; u += 128)
      copy16_to_lds(ks + u, (uint4*)s_k[0] + u);
    for (int u = threadIdx.x; u < 512; u += 128) {
      int c = u >> 3, q4 = u & 7;
      copy16_to_lds((const uint4*)(vbp + (size_t)c*4096) + q4, (uint4*)(s_v[0] + c*64) + q4);
    }
  }
  async_fence();
  __syncthreads();

  AB16 aq0, aq1;                                  // Q as A operand, K-dim = c
  #pragma unroll
  for (int i = 0; i < 8; ++i) {
    int kb = a_kbase(i, hi);
    aq0.u[i] = *(const uint32_t*)(s_q + (wave*16 + lo)*64 + kb);
    aq1.u[i] = *(const uint32_t*)(s_q + (wave*16 + lo)*64 + kb + 32);
  }
  _Float16* pw = s_q + wave*16*64;                // per-wave P scratch (own Q rows)

  float mrun[8], lrun[8];
  ACC8 o[4];
  #pragma unroll
  for (int i = 0; i < 8; ++i) { mrun[i] = -3.0e38f; lrun[i] = 0.f; }
  #pragma unroll
  for (int ct = 0; ct < 4; ++ct)
    #pragma unroll
    for (int e = 0; e < 8; ++e) o[ct].f[e] = 0.f;

  int p = 0;
  for (int m0 = 0; m0 < 4096; m0 += 64) {
    // issue next tile's staging into the other buffer
    if (m0 + 64 < 4096) {
      int np = 1 - p;
      const uint4* ks = (const uint4*)(kbp + (size_t)(m0 + 64)*64);
      for (int u = threadIdx.x; u < 512; u += 128)
        copy16_to_lds(ks + u, (uint4*)s_k[np] + u);
      for (int u = threadIdx.x; u < 512; u += 128) {
        int c = u >> 3, q4 = u & 7;
        copy16_to_lds((const uint4*)(vbp + (size_t)c*4096 + m0 + 64) + q4,
                      (uint4*)(s_v[np] + c*64) + q4);
      }
    }
    const _Float16* kb_t = s_k[p];
    const _Float16* vb_t = s_v[p];

    // S = (Q^T K) * C^-0.5   (4 key sub-tiles of 16, K-dim c split 2x32)
    ACC8 s[4];
    #pragma unroll
    for (int t = 0; t < 4; ++t) {
      AB16 b0, b1;
      int row = t*16 + lo;                       // key token; B pairs = (c,c+1)
      #pragma unroll
      for (int i = 0; i < 8; ++i) {
        int c0 = hi*16 + 2*i;
        b0.u[i] = *(const uint32_t*)(kb_t + row*64 + c0);
        b1.u[i] = *(const uint32_t*)(kb_t + row*64 + 32 + c0);
      }
      ACC8 z;
      #pragma unroll
      for (int e = 0; e < 8; ++e) z.f[e] = 0.f;
      z.v = WMMA16(aq0.v, b0.v, z.v);
      z.v = WMMA16(aq1.v, b1.v, z.v);
      #pragma unroll
      for (int e = 0; e < 8; ++e) s[t].f[e] = z.f[e] * 0.125f;
    }

    // online softmax (rows live in 16-lane half-groups per C/D layout)
    #pragma unroll
    for (int i = 0; i < 8; ++i) {
      float mx = fmaxf(fmaxf(s[0].f[i], s[1].f[i]), fmaxf(s[2].f[i], s[3].f[i]));
      #pragma unroll
      for (int d = 1; d <= 8; d <<= 1) mx = fmaxf(mx, __shfl_xor(mx, d, 32));
      float mnew = fmaxf(mrun[i], mx);
      float rs = 0.f;
      #pragma unroll
      for (int t = 0; t < 4; ++t) {
        float pr = __expf(s[t].f[i] - mnew); s[t].f[i] = pr; rs += pr;
      }
      #pragma unroll
      for (int d = 1; d <= 8; d <<= 1) rs += __shfl_xor(rs, d, 32);
      float alpha = __expf(mrun[i] - mnew);
      lrun[i] = lrun[i]*alpha + rs;
      mrun[i] = mnew;
      #pragma unroll
      for (int ct = 0; ct < 4; ++ct) o[ct].f[i] *= alpha;
    }

    // P: C/D layout -> row-major LDS -> A layout (per-wave scratch)
    #pragma unroll
    for (int t = 0; t < 4; ++t)
      #pragma unroll
      for (int i = 0; i < 8; ++i)
        pw[(i + 8*hi)*64 + t*16 + lo] = (_Float16)s[t].f[i];
    asm volatile("s_wait_dscnt 0" ::: "memory");
    AB16 ap0, ap1;
    #pragma unroll
    for (int i = 0; i < 8; ++i) {
      int kb = a_kbase(i, hi);
      ap0.u[i] = *(const uint32_t*)(pw + lo*64 + kb);
      ap1.u[i] = *(const uint32_t*)(pw + lo*64 + kb + 32);
    }

    // O += P x V^T   (B[K=m][N=c] = s_v[c][m], pairs = (m,m+1) contiguous)
    #pragma unroll
    for (int ct = 0; ct < 4; ++ct) {
      AB16 b0, b1;
      int cidx = ct*16 + lo;
      #pragma unroll
      for (int i = 0; i < 8; ++i) {
        int mp = hi*16 + 2*i;
        b0.u[i] = *(const uint32_t*)(vb_t + cidx*64 + mp);
        b1.u[i] = *(const uint32_t*)(vb_t + cidx*64 + 32 + mp);
      }
      o[ct].v = WMMA16(ap0.v, b0.v, o[ct].v);
      o[ct].v = WMMA16(ap1.v, b1.v, o[ct].v);
    }

    async_fence();        // next tile resident in LDS
    __syncthreads();      // everyone done with buf p / buf np fully written
    p = 1 - p;
  }

  // normalize, stage [c][n_local], store channel-major attn[b][c][n]
  #pragma unroll
  for (int i = 0; i < 8; ++i) {
    float inv = 1.f / lrun[i];
    #pragma unroll
    for (int ct = 0; ct < 4; ++ct)
      s_q[(ct*16 + lo)*64 + wave*16 + i + 8*hi] = (_Float16)(o[ct].f[i] * inv);
  }
  __syncthreads();
  for (int u = threadIdx.x; u < 512; u += 128) {
    int c = u >> 3, q4 = u & 7;
    ((uint4*)(attn + ((size_t)(b*64 + c))*4096 + qbase))[q4] = ((const uint4*)s_q)[u];
  }
}

// ---------------------------------------------------------------------------
// Kernel 4: proj linear on last axis + bias + residual -> out (f32)
// ---------------------------------------------------------------------------
__global__ __launch_bounds__(128) void proj_kernel(const _Float16* __restrict__ attn,
    const _Float16* __restrict__ wh, const float* __restrict__ bp,
    const float* __restrict__ x, float* __restrict__ out) {
  int b = blockIdx.x >> 6, h = blockIdx.x & 63;
  int lane = threadIdx.x & 31, wave = threadIdx.x >> 5;
  int lo = lane & 15, hi = lane >> 4;
  const _Float16* Wh = wh + 12288;       // f16 Wp
  const _Float16* arow = attn + ((size_t)(b*64 + wave*16 + lo))*4096 + h*64;
  AB16 a0, a1;
  #pragma unroll
  for (int i = 0; i < 8; ++i) {
    int kb = a_kbase(i, hi);
    a0.u[i] = *(const uint32_t*)(arow + kb);
    a1.u[i] = *(const uint32_t*)(arow + kb + 32);
  }
  #pragma unroll
  for (int jt = 0; jt < 4; ++jt) {
    int j = jt*16 + lo;
    AB16 b0, b1;
    #pragma unroll
    for (int i = 0; i < 8; ++i) {
      int c0 = hi*16 + 2*i;
      b0.u[i] = *(const uint32_t*)(Wh + j*64 + c0);
      b1.u[i] = *(const uint32_t*)(Wh + j*64 + 32 + c0);
    }
    ACC8 acc;
    #pragma unroll
    for (int e = 0; e < 8; ++e) acc.f[e] = 0.f;
    acc.v = WMMA16(a0.v, b0.v, acc.v);
    acc.v = WMMA16(a1.v, b1.v, acc.v);
    float bj = bp[j];
    #pragma unroll
    for (int i = 0; i < 8; ++i) {
      int c = wave*16 + i + 8*hi;
      size_t addr = (((size_t)(b*64 + c))*64 + h)*64 + j;
      out[addr] = acc.f[i] + bj + x[addr];
    }
  }
}

// ---------------------------------------------------------------------------
extern "C" void kernel_launch(void* const* d_in, const int* in_sizes, int n_in,
                              void* d_out, int out_size, void* d_ws, size_t ws_size,
                              hipStream_t stream) {
  (void)in_sizes; (void)n_in; (void)out_size; (void)ws_size;
  const float* x  = (const float*)d_in[0];
  const float* gw = (const float*)d_in[1];
  const float* gb = (const float*)d_in[2];
  const float* Wq = (const float*)d_in[3];
  const float* bq = (const float*)d_in[4];
  const float* Wk = (const float*)d_in[5];
  const float* bk = (const float*)d_in[6];
  const float* Wv = (const float*)d_in[7];
  const float* bv = (const float*)d_in[8];
  const float* Wp = (const float*)d_in[9];
  const float* bp = (const float*)d_in[10];
  float* out = (float*)d_out;

  const size_t elems = (size_t)8 * 64 * 64 * 64;   // 2,097,152
  _Float16* xn = (_Float16*)d_ws;                  // 4 MiB (reused as attn out)
  _Float16* qT = xn + elems;                       // 4 MiB, [b][n][c]
  _Float16* kT = qT + elems;                       // 4 MiB, [b][n][c]
  _Float16* vN = kT + elems;                       // 4 MiB, [b][c][n]
  _Float16* wh = vN + elems;                       // 32 KiB f16 weights
  _Float16* attn = xn;                             // reuse xn after QKV

  prep_weights<<<dim3(16),  dim3(256), 0, stream>>>(Wq, Wk, Wv, Wp, wh);
  gn_kernel   <<<dim3(256), dim3(256), 0, stream>>>(x, gw, gb, xn);
  qkv_kernel  <<<dim3(512), dim3(128), 0, stream>>>(xn, wh, bq, bk, bv, qT, kT, vN);
  attn_kernel <<<dim3(512), dim3(128), 0, stream>>>(qT, kT, vN, attn);
  proj_kernel <<<dim3(512), dim3(128), 0, stream>>>(attn, wh, bp, x, out);
}
